// AutoCorrelation_80255758893093
// MI455X (gfx1250) — matrix operand order repaired and checked by reading, not compiled, not run on hardware
//
#include <hip/hip_runtime.h>

// MI455X / gfx1250, wave32. AutoCorrelation: B=4,H=8,L=4096,Dh=64, k_top=8.
//
// corr[tau] = (1/64) sum_t sum_d q[t,d]*k[(t-tau)%L,d] done as WMMA tiles:
//   T_B[i,j] = sum_c sum_d Q[16c+i,d] * K[(16(c-B)+j)%L,d]   (lag = 16B+i-j)
// then diagonal assembly, top-8 + softmax, weighted circular gather of v.

typedef __attribute__((ext_vector_type(16))) _Float16 v16h;
typedef __attribute__((ext_vector_type(8)))  float    v8f;
typedef __attribute__((ext_vector_type(4)))  unsigned int u32x4;
typedef __attribute__((ext_vector_type(4)))  float    f32x4;
typedef int gvi4 __attribute__((vector_size(16)));   // matches builtin's V4i

#define L      4096
#define DH     64
#define NBH    32      // B*H
#define NTILE  256     // L/16 lag tiles per head
#define KTOP   8       // int(log(4097)) = 8
#define KSLOTS 16      // K sliding-window slots (16-row blocks)

#define AS1 __attribute__((address_space(1)))
#define AS3 __attribute__((address_space(3)))

#if defined(__has_builtin)
#if __has_builtin(__builtin_amdgcn_global_load_async_to_lds_b128)
#define HAVE_ASYNC_LDS 1
#endif
#endif
#ifndef HAVE_ASYNC_LDS
#define HAVE_ASYNC_LDS 0
#endif

union Frag  { u32x4 u[2]; v16h v; };     // 32B WMMA operand (16 halfs/lane)
union Pack8 { _Float16 h[8]; u32x4 u; }; // 8 halfs = 16B chunk

__device__ inline u32x4 pack_f16(f32x4 a, f32x4 b) {
  Pack8 p;
  p.h[0] = (_Float16)a.x; p.h[1] = (_Float16)a.y;
  p.h[2] = (_Float16)a.z; p.h[3] = (_Float16)a.w;
  p.h[4] = (_Float16)b.x; p.h[5] = (_Float16)b.y;
  p.h[6] = (_Float16)b.z; p.h[7] = (_Float16)b.w;
  return p.u;
}

__device__ inline void wait_async0() {
#if HAVE_ASYNC_LDS
#if __has_builtin(__builtin_amdgcn_s_wait_asynccnt)
  __builtin_amdgcn_s_wait_asynccnt(0);
#else
  asm volatile("s_wait_asynccnt 0" ::: "memory");
#endif
#endif
}

#if HAVE_ASYNC_LDS
__device__ inline void async_cp16(const _Float16* g, _Float16* l) {
  __builtin_amdgcn_global_load_async_to_lds_b128(
      (AS1 gvi4*)(void*)g, (AS3 gvi4*)(void*)l, 0, 0);
}
#endif

// ---------------------------------------------------------------------------
// Stage 0: one-shot f32 -> f16 conversion of q and k into workspace.
// ---------------------------------------------------------------------------
__global__ __launch_bounds__(256)
void cvt_f16_kernel(const float* __restrict__ q, const float* __restrict__ k,
                    _Float16* __restrict__ qh, _Float16* __restrict__ kh) {
  const size_t i = ((size_t)blockIdx.x * 256 + threadIdx.x) * 8;
  f32x4 a = *(const f32x4*)&q[i];
  f32x4 b = *(const f32x4*)&q[i + 4];
  *(u32x4*)&qh[i] = pack_f16(a, b);
  a = *(const f32x4*)&k[i];
  b = *(const f32x4*)&k[i + 4];
  *(u32x4*)&kh[i] = pack_f16(a, b);
}

// ---------------------------------------------------------------------------
// Stage 1: WMMA lag tiles. One WG (8 waves) per (head, 8 lag tiles).
// LDS: 16-slot circular window of 16-row K blocks (32KB) + 2 Q blocks (4KB).
// Async global->LDS copies (ASYNCcnt) overlap the WMMA compute.
// ---------------------------------------------------------------------------
__global__ __launch_bounds__(256)
void corr_tiles_kernel(const _Float16* __restrict__ qh,
                       const _Float16* __restrict__ kh,
                       float* __restrict__ tiles) {
  __shared__ __align__(16) _Float16 sK[KSLOTS * 16 * DH];  // 32 KB
  __shared__ __align__(16) _Float16 sQ[2 * 16 * DH];       // 4 KB (double buf)

  const int bh  = blockIdx.x;
  const int B0  = blockIdx.y * 8;          // first lag tile of this WG
  const int tid = threadIdx.x;
  const int ln  = tid & 31;                // lane in wave32
  const int wv  = tid >> 5;                // wave id -> lag tile B0+wv
  const int hi  = (ln >> 4) & 1;           // lane half-group
  const int mn  = ln & 15;                 // A row i  /  B col j

  const size_t base = (size_t)bh * L * DH;
  const _Float16* qg = qh + base;
  const _Float16* kg = kh + base;

  // ---- preload: Q block 0 (-> qslot 0) + K blocks (-B0-7 .. -B0) ----
  for (int chunk = tid; chunk < 128 + 8 * 128; chunk += 256) {
    const _Float16* src;
    _Float16* dst;
    if (chunk < 128) {                       // waves 0-3 on first round
      const int row = chunk >> 3;
      const int d   = (chunk & 7) * 8;
      src = &qg[row * DH + d];
      dst = &sQ[row * DH + d];
    } else {
      const int cc   = chunk - 128;
      const int blk  = cc >> 7;              // 0..7
      const int nb   = blk - 7 - B0;         // logical K block index
      const int c2   = cc & 127;
      const int row  = c2 >> 3;
      const int d    = (c2 & 7) * 8;
      const int grow = (16 * nb + row) & (L - 1);
      src = &kg[grow * DH + d];
      dst = &sK[((nb & (KSLOTS - 1)) * 16 + row) * DH + d];
    }
#if HAVE_ASYNC_LDS
    async_cp16(src, dst);
#else
    *(u32x4*)dst = *(const u32x4*)src;
#endif
  }
  wait_async0();
  __syncthreads();

  v8f acc = {0.f, 0.f, 0.f, 0.f, 0.f, 0.f, 0.f, 0.f};

  for (int c = 0; c < NTILE; ++c) {
#if HAVE_ASYNC_LDS
    // Issue DMA for step c+1 at the top; target slots are disjoint from the
    // step-c read set and (thanks to last iteration's barrier) from step c-1.
    if (c < NTILE - 1) {
      if (tid < 128) {                       // waves 0-3: next Q block
        const int row = tid >> 3;
        const int d   = (tid & 7) * 8;
        async_cp16(&qg[(16 * (c + 1) + row) * DH + d],
                   &sQ[((c + 1) & 1) * (16 * DH) + row * DH + d]);
      } else {                               // waves 4-7: next K block
        const int cc   = tid - 128;
        const int row  = cc >> 3;
        const int d    = (cc & 7) * 8;
        const int nb   = c + 1 - B0;
        const int grow = (16 * nb + row) & (L - 1);
        async_cp16(&kg[grow * DH + d],
                   &sK[((nb & (KSLOTS - 1)) * 16 + row) * DH + d]);
      }
    }
#else
    u32x4 st = {0u, 0u, 0u, 0u};
    int ldsOff = 0;
    if (c < NTILE - 1) {
      if (tid < 128) {
        const int row = tid >> 3;
        const int d   = (tid & 7) * 8;
        st = *(const u32x4*)&qg[(16 * (c + 1) + row) * DH + d];
        ldsOff = ((c + 1) & 1) * (16 * DH) + row * DH + d;
      } else {
        const int cc   = tid - 128;
        const int row  = cc >> 3;
        const int d    = (cc & 7) * 8;
        const int nb   = c + 1 - B0;
        const int grow = (16 * nb + row) & (L - 1);
        st = *(const u32x4*)&kg[grow * DH + d];
        ldsOff = -(((nb & (KSLOTS - 1)) * 16 + row) * DH + d) - 1; // K marker
      }
    }
#endif

    // ---- compute: 2 x (16x16x32 f16) WMMA over Dh = 64 ----
    // A 16x32 f16: lane ln: M=ln&15; halves0-7 -> K=8*hi+0..7,
    //                               halves8-15 -> K=16+8*hi+0..7.
    // B 32x16 f16: lane ln: N=ln&15; halves0-7 -> K=8*hi+0..7,
    //                               halves8-15 -> K=16+8*hi+0..7.
    {
      const int slot = (c - B0 - wv) & (KSLOTS - 1);
      const _Float16* qb = sQ + (c & 1) * (16 * DH);
      const _Float16* kb = sK + slot * (16 * DH);
      Frag A, Bm;
      A.u[0]  = *(const u32x4*)&qb[mn * DH + 0  + 8 * hi];
      A.u[1]  = *(const u32x4*)&qb[mn * DH + 16 + 8 * hi];
      Bm.u[0] = *(const u32x4*)&kb[mn * DH + 0  + 8 * hi];
      Bm.u[1] = *(const u32x4*)&kb[mn * DH + 16 + 8 * hi];
      acc = __builtin_amdgcn_wmma_f32_16x16x32_f16(false, A.v, false, Bm.v,
                                                   (short)0, acc, false, false);
      A.u[0]  = *(const u32x4*)&qb[mn * DH + 32 + 8 * hi];
      A.u[1]  = *(const u32x4*)&qb[mn * DH + 48 + 8 * hi];
      Bm.u[0] = *(const u32x4*)&kb[mn * DH + 32 + 8 * hi];
      Bm.u[1] = *(const u32x4*)&kb[mn * DH + 48 + 8 * hi];
      acc = __builtin_amdgcn_wmma_f32_16x16x32_f16(false, A.v, false, Bm.v,
                                                   (short)0, acc, false, false);
    }

#if HAVE_ASYNC_LDS
    wait_async0();       // this wave's DMA for step c+1 landed in LDS
    __syncthreads();     // publish across waves; protects slot reuse
#else
    __syncthreads();     // everyone done reading LDS
    if (c < NTILE - 1) {
      if (tid < 128) *(u32x4*)&sQ[ldsOff] = st;
      else           *(u32x4*)&sK[-(ldsOff + 1)] = st;
    }
    __syncthreads();     // LDS ready for next iteration
#endif
  }

  // Store tile: C/D layout -> element (M = r+8*hi, N = ln&15) in VGPR r.
  float* dst = tiles + (size_t)(bh * NTILE + (B0 + wv)) * 256;
#pragma unroll
  for (int r = 0; r < 8; ++r)
    dst[(r + 8 * hi) * 16 + mn] = acc[r];
}

// ---------------------------------------------------------------------------
// Stage 2: corr[16B+d] = sum_{i>=d} T_B[i,i-d] + sum_{i<d} T_{B+1}[i,i-d+16],
// scaled by 1/Dh (reference takes mean over Dh).
// ---------------------------------------------------------------------------
__global__ __launch_bounds__(256)
void corr_assemble_kernel(const float* __restrict__ tiles, float* __restrict__ corr) {
  const int bh  = blockIdx.x;
  const int tau = blockIdx.y * 256 + threadIdx.x;
  const int Bt  = tau >> 4;
  const int dlt = tau & 15;
  const float* t0 = tiles + (size_t)(bh * NTILE + Bt) * 256;
  const float* t1 = tiles + (size_t)(bh * NTILE + ((Bt + 1) & (NTILE - 1))) * 256;
  float s = 0.f;
  for (int i = dlt; i < 16; ++i) s += t0[i * 16 + (i - dlt)];
  for (int i = 0; i < dlt; ++i)  s += t1[i * 16 + (i - dlt + 16)];
  corr[bh * L + tau] = s * (1.0f / DH);
}

// ---------------------------------------------------------------------------
// Stage 3: per head, top-8 lags (ties -> lower index, like lax.top_k) + softmax.
// ---------------------------------------------------------------------------
__global__ __launch_bounds__(256)
void topk_softmax_kernel(const float* __restrict__ corr,
                         float* __restrict__ attn, int* __restrict__ delays) {
  __shared__ float sc[L];
  __shared__ float rv[256];
  __shared__ int   ri[256];
  __shared__ float wsel[KTOP];
  __shared__ int   dsel[KTOP];
  const int bh  = blockIdx.x;
  const int tid = threadIdx.x;
  for (int t = tid; t < L; t += 256) sc[t] = corr[bh * L + t];
  __syncthreads();

  for (int p = 0; p < KTOP; ++p) {
    float bv = -3.402823466e38f; int bi = L;
    for (int t = tid; t < L; t += 256) {
      const float vv = sc[t];
      if (vv > bv || (vv == bv && t < bi)) { bv = vv; bi = t; }
    }
    rv[tid] = bv; ri[tid] = bi;
    __syncthreads();
    for (int s = 128; s > 0; s >>= 1) {
      if (tid < s) {
        const float ov = rv[tid + s]; const int oi = ri[tid + s];
        if (ov > rv[tid] || (ov == rv[tid] && oi < ri[tid])) { rv[tid] = ov; ri[tid] = oi; }
      }
      __syncthreads();
    }
    if (tid == 0) { wsel[p] = rv[0]; dsel[p] = ri[0]; sc[ri[0]] = -3.402823466e38f; }
    __syncthreads();
  }

  if (tid == 0) {
    const float m = wsel[0];
    float e[KTOP], esum = 0.f;
    for (int j = 0; j < KTOP; ++j) { e[j] = expf(wsel[j] - m); esum += e[j]; }
    for (int j = 0; j < KTOP; ++j) {
      attn[bh * KTOP + j]   = e[j] / esum;
      delays[bh * KTOP + j] = dsel[j];
    }
  }
}

// ---------------------------------------------------------------------------
// Stage 4: out[t,d] = sum_j attn_j * v[(t - delay_j) mod L, d]  (float4/lane)
// ---------------------------------------------------------------------------
__global__ __launch_bounds__(256)
void gather_out_kernel(const float* __restrict__ v, const float* __restrict__ attn,
                       const int* __restrict__ delays, float* __restrict__ out) {
  const int bh  = blockIdx.x;
  const int tid = threadIdx.x;
  const int t   = blockIdx.y * 16 + (tid >> 4);
  const int d   = (tid & 15) * 4;
  const float* vb = v   + (size_t)bh * L * DH;
  float*       ob = out + (size_t)bh * L * DH;
  f32x4 acc = {0.f, 0.f, 0.f, 0.f};
#pragma unroll
  for (int j = 0; j < KTOP; ++j) {
    const int   dl = delays[bh * KTOP + j];
    const float a  = attn[bh * KTOP + j];
    const int   s  = (t - dl) & (L - 1);
    const f32x4 vv = *(const f32x4*)&vb[s * DH + d];
    acc += vv * a;
  }
  *(f32x4*)&ob[t * DH + d] = acc;
}

// ---------------------------------------------------------------------------
extern "C" void kernel_launch(void* const* d_in, const int* in_sizes, int n_in,
                              void* d_out, int out_size, void* d_ws, size_t ws_size,
                              hipStream_t stream) {
  const float* q = (const float*)d_in[0];
  const float* k = (const float*)d_in[1];
  const float* v = (const float*)d_in[2];
  float* out = (float*)d_out;

  const size_t nelem = (size_t)NBH * L * DH;   // 8,388,608 per tensor

  // Workspace: qh f16 (16MB) | kh f16 (16MB) | tiles f32 (8MB) | corr | attn | delays
  _Float16* qh   = (_Float16*)d_ws;
  _Float16* kh   = qh + nelem;
  float*    tiles = (float*)(kh + nelem);
  float*    corr  = tiles + (size_t)NBH * NTILE * 256;
  float*    attn  = corr + (size_t)NBH * L;
  int*      delays = (int*)(attn + NBH * KTOP);

  cvt_f16_kernel      <<<dim3(nelem / (256 * 8)), 256, 0, stream>>>(q, k, qh, kh);
  corr_tiles_kernel   <<<dim3(NBH, NTILE / 8), 256, 0, stream>>>(qh, kh, tiles);
  corr_assemble_kernel<<<dim3(NBH, L / 256),   256, 0, stream>>>(tiles, corr);
  topk_softmax_kernel <<<dim3(NBH),            256, 0, stream>>>(corr, attn, delays);
  gather_out_kernel   <<<dim3(NBH, L / 16),    256, 0, stream>>>(v, attn, delays, out);
}
